// ICLAttention_29257317220688
// MI455X (gfx1250) — matrix operand order repaired and checked by reading, not compiled, not run on hardware
//
#include <hip/hip_runtime.h>
#include <hip/hip_bf16.h>

#define BB 2
#define SS 2048
#define EE 2048
#define HH 16
#define DD 128
#define MM (BB*SS) /* 4096 */

// padded LDS row strides (bf16 elems) for bank-conflict-free fragment reads
#define KLD 136   /* 128 + 8 */
#define VLD 72    /* 64 + 8  */
#define PLD 72
#define GLD 72    /* gemm k-tile stride: 64 + 8 */

typedef __attribute__((ext_vector_type(16))) __bf16 v16bf;
typedef __attribute__((ext_vector_type(8)))  __bf16 v8bf;
typedef __attribute__((ext_vector_type(8)))  float  v8f;

#if __has_builtin(__builtin_amdgcn_sched_barrier)
#define SCHED_BARRIER() __builtin_amdgcn_sched_barrier(0)
#else
#define SCHED_BARRIER()
#endif

__device__ __forceinline__ __bf16 f2bf(float f) {
  unsigned u = __builtin_bit_cast(unsigned, f);
  u += 0x7FFFu + ((u >> 16) & 1u);           // round-to-nearest-even
  unsigned short h = (unsigned short)(u >> 16);
  return __builtin_bit_cast(__bf16, h);
}

// ---- WMMA fragment loaders (CDNA5 16x16x32 bf16 VGPR layouts) ----

// A matrix 16x32 (MxK), row-major source; ptr at (m0,k0).
__device__ __forceinline__ v16bf load_a(const __bf16* __restrict__ A, int lda, int lane) {
  int ml = lane & 15, hi = lane >> 4;
  const __bf16* p = A + (size_t)ml * lda + hi * 16;
  v16bf a;
#pragma unroll
  for (int e = 0; e < 16; ++e) a[e] = p[e];
  return a;
}

// B matrix 32x16 (KxN) where memory holds the N-major / K-contiguous form (src[n*ld + k]).
__device__ __forceinline__ v16bf load_b_colmajor(const __bf16* __restrict__ Wt, int ld, int lane) {
  int nl = lane & 15, hi = lane >> 4;
  const __bf16* p = Wt + (size_t)nl * ld + hi * 16;
  v16bf b;
#pragma unroll
  for (int e = 0; e < 16; ++e) b[e] = p[e];
  return b;
}

__device__ __forceinline__ v8f wmma_bf16(v16bf a, v16bf b, v8f c) {
  return __builtin_amdgcn_wmma_f32_16x16x32_bf16(false, a, false, b, (short)0, c, false, false);
}

// ---- fp32 -> bf16 conversion ----
__global__ void cvt_bf16(const float* __restrict__ src, __bf16* __restrict__ dst, size_t n) {
  size_t i = (size_t)blockIdx.x * blockDim.x + threadIdx.x;
  if (i < n) dst[i] = f2bf(src[i]);
}

// ---- GEMM: C[M,N] = A[M,K] @ W[N,K]^T, bf16 in, fp32 accumulate ----
// LDS-staged, register-pipelined. Block 256 thr (8 waves 2x4), block tile 128x256,
// wave tile 64x64, k-step 64 (two 32-wide WMMA substeps from LDS).
// MODE 0: store bf16 V^T to [B,H,D,S], vectorized 16B stores   (V projection)
// MODE 1: RoPE epilogue, store bf16 [B,H,S,D]                  (Q/K projections)
// MODE 2: store fp32 row-major [M,N]                           (output projection)
template <int MODE>
__global__ __launch_bounds__(256) void gemm_qkvo(const __bf16* __restrict__ A,
                                                 const __bf16* __restrict__ W,
                                                 __bf16* __restrict__ outb,
                                                 float* __restrict__ outf,
                                                 float oscale) {
  __shared__ __bf16 Alds[128 * GLD];  // 18KB: [m][k], padded
  __shared__ __bf16 Wlds[256 * GLD];  // 36KB: [n][k], padded
  const int tid = threadIdx.x;
  const int lane = tid & 31;
  const int wave = tid >> 5;
  const int wm = wave >> 2, wn = wave & 3;     // 2x4 wave grid
  const int mB = blockIdx.y * 128;
  const int nB = blockIdx.x * 256;
  const int m0 = mB + wm * 64;
  const int n0 = nB + wn * 64;

  const __bf16* pA = A + (size_t)mB * EE;
  const __bf16* pW = W + (size_t)nB * EE;

  // prologue: stage k-tile 0.  A: 128 rows x 8 16B-chunks; W: 256 rows x 8 chunks.
#pragma unroll
  for (int i = 0; i < 4; ++i) {
    int c = tid + i * 256;
    *(v8bf*)(Alds + (size_t)(c >> 3) * GLD + (c & 7) * 8) =
        *(const v8bf*)(pA + (size_t)(c >> 3) * EE + (c & 7) * 8);
  }
#pragma unroll
  for (int i = 0; i < 8; ++i) {
    int c = tid + i * 256;
    *(v8bf*)(Wlds + (size_t)(c >> 3) * GLD + (c & 7) * 8) =
        *(const v8bf*)(pW + (size_t)(c >> 3) * EE + (c & 7) * 8);
  }
  __syncthreads();

  v8f acc[4][4] = {};
  for (int k0 = 0; k0 < EE; k0 += 64) {
    // 1) issue next k-tile's global loads; latency hidden by 32 WMMAs below
    v8bf ast[4], wst[8];
    const bool havenext = (k0 + 64) < EE;
    if (havenext) {
#pragma unroll
      for (int i = 0; i < 4; ++i) {
        int c = tid + i * 256;
        ast[i] = *(const v8bf*)(pA + (size_t)(c >> 3) * EE + k0 + 64 + (c & 7) * 8);
      }
#pragma unroll
      for (int i = 0; i < 8; ++i) {
        int c = tid + i * 256;
        wst[i] = *(const v8bf*)(pW + (size_t)(c >> 3) * EE + k0 + 64 + (c & 7) * 8);
      }
    }
    SCHED_BARRIER();  // pin the loads above this point

    // 2) 32 WMMAs from LDS (two 32-wide substeps)
#pragma unroll
    for (int ks = 0; ks < 2; ++ks) {
      v16bf a[4], b[4];
#pragma unroll
      for (int i = 0; i < 4; ++i)
        a[i] = load_a(Alds + (size_t)(wm * 64 + 16 * i) * GLD + ks * 32, GLD, lane);
#pragma unroll
      for (int j = 0; j < 4; ++j)
        b[j] = load_b_colmajor(Wlds + (size_t)(wn * 64 + 16 * j) * GLD + ks * 32, GLD, lane);
#pragma unroll
      for (int i = 0; i < 4; ++i)
#pragma unroll
        for (int j = 0; j < 4; ++j)
          acc[i][j] = wmma_bf16(a[i], b[j], acc[i][j]);
    }

    __syncthreads();  // all waves done reading this k-tile
    if (havenext) {
#pragma unroll
      for (int i = 0; i < 4; ++i) {
        int c = tid + i * 256;
        *(v8bf*)(Alds + (size_t)(c >> 3) * GLD + (c & 7) * 8) = ast[i];
      }
#pragma unroll
      for (int i = 0; i < 8; ++i) {
        int c = tid + i * 256;
        *(v8bf*)(Wlds + (size_t)(c >> 3) * GLD + (c & 7) * 8) = wst[i];
      }
    }
    __syncthreads();  // next k-tile visible
  }

  const int nl = lane & 15, hi = lane >> 4;
#pragma unroll
  for (int i = 0; i < 4; ++i) {
#pragma unroll
    for (int j = 0; j < 4; ++j) {
      int n = n0 + 16 * j + nl;
      if (MODE == 2) {
#pragma unroll
        for (int r = 0; r < 8; ++r) {
          int m = m0 + 16 * i + r + 8 * hi;
          outf[(size_t)m * EE + n] = acc[i][j][r];
        }
      } else if (MODE == 0) {
        // V^T [B,H,D,S]: lane holds 8 consecutive s for fixed d -> one 16B store
        int h = n >> 7, d = n & (DD - 1);
        int mb = m0 + 16 * i + 8 * hi;
        int bidx = mb >> 11;
        int sb = mb & (SS - 1);
        v8bf pk;
#pragma unroll
        for (int r = 0; r < 8; ++r) pk[r] = f2bf(acc[i][j][r] * oscale);
        size_t off = (((size_t)(bidx * HH + h)) * DD + d) * SS + sb;
        *(v8bf*)(outb + off) = pk;
      } else {
        int h = n >> 7, d = n & (DD - 1);
        float invf = __expf(-((float)(d & ~1) / (float)DD) * 9.210340371976184f);  // 10000^(-(d&~1)/D)
#pragma unroll
        for (int r = 0; r < 8; ++r) {
          int m = m0 + 16 * i + r + 8 * hi;
          int s = m & (SS - 1);
          float v = acc[i][j][r];
          float ang = (float)s * invf;
          float c = cosf(ang), sn = sinf(ang);
          float pv = __shfl_xor(v, 1, 32);  // rotary partner column (d ^ 1)
          v = (d & 1) ? (v * c + pv * sn) : (v * c - pv * sn);
          v *= oscale;
          int bidx = m >> 11;  // m / S
          size_t off = (((size_t)(bidx * HH + h)) * SS + (size_t)s) * DD + d;
          outb[off] = f2bf(v);
        }
      }
    }
  }
}

// ---- flash attention, LDS-staged + register-pipelined ----
// Block: 128 threads (4 waves); wave owns 16 query rows x full D=128; 64-key tiles.
// Q,K in [B,H,S,D]; V transposed in [B,H,D,S]. Padded LDS rows: conflict-free frag reads.
__global__ __launch_bounds__(128) void attn_fa(const __bf16* __restrict__ Qh,
                                               const __bf16* __restrict__ Kh,
                                               const __bf16* __restrict__ Vt,
                                               __bf16* __restrict__ O) {
  __shared__ __bf16 Klds[64 * KLD];    // [key][d], padded
  __shared__ __bf16 Vlds[128 * VLD];   // [d][key], padded
  __shared__ __bf16 pbuf[4][16 * PLD]; // per-wave P staging, padded
  const int tid = threadIdx.x;
  const int lane = tid & 31;
  const int wave = tid >> 5;
  const int bh = blockIdx.y;
  const int b = bh >> 4, h = bh & 15;
  const int q0 = blockIdx.x * 64 + wave * 16;
  const __bf16* Qb = Qh + (size_t)bh * SS * DD;
  const __bf16* Kb = Kh + (size_t)bh * SS * DD;
  const __bf16* Vb = Vt + (size_t)bh * DD * SS;
  const int nl = lane & 15, hi = lane >> 4;

  // stage tile 0.  K tile: 128 rows x 8 16B-chunks; V tile: 128 rows x 8 chunks.
#pragma unroll
  for (int i = 0; i < 8; ++i) {
    int c = tid + i * 128;
    *(v8bf*)(Klds + (size_t)(c >> 3) * KLD + (c & 7) * 8) =
        *(const v8bf*)(Kb + (size_t)(c >> 3) * DD + (c & 7) * 8);
  }
#pragma unroll
  for (int i = 0; i < 8; ++i)
    *(v8bf*)(Vlds + (size_t)tid * VLD + i * 8) = *(const v8bf*)(Vb + (size_t)tid * SS + i * 8);

  v16bf aQ[4];  // Q already carries the 1/sqrt(E) scale from the projection
#pragma unroll
  for (int ks = 0; ks < 4; ++ks)
    aQ[ks] = load_a(Qb + (size_t)q0 * DD + ks * 32, DD, lane);

  v8f o[8] = {};
  float mrow[8], lrow[8];
#pragma unroll
  for (int r = 0; r < 8; ++r) { mrow[r] = -1e30f; lrow[r] = 0.f; }

  __syncthreads();

  for (int t0 = 0; t0 < SS; t0 += 64) {
    // 1) issue next tile's global loads now; drain into LDS at end of iteration
    v8bf kst[8], vst[8];
    const bool havenext = (t0 + 64) < SS;
    if (havenext) {
      const __bf16* ksrc = Kb + (size_t)(t0 + 64) * DD;
#pragma unroll
      for (int i = 0; i < 8; ++i) {
        int c = tid + i * 128;
        kst[i] = *(const v8bf*)(ksrc + (size_t)(c >> 3) * DD + (c & 7) * 8);
      }
      const __bf16* vsrc = Vb + (size_t)tid * SS + t0 + 64;
#pragma unroll
      for (int i = 0; i < 8; ++i) vst[i] = *(const v8bf*)(vsrc + i * 8);
    }
    SCHED_BARRIER();  // pin the loads above this point

    // 2) scores: S[16x64] = Qs @ K^T from LDS (16 WMMAs)
    v8f sc[4] = {};
#pragma unroll
    for (int ks = 0; ks < 4; ++ks) {
      v16bf bK[4];
#pragma unroll
      for (int j = 0; j < 4; ++j)
        bK[j] = load_b_colmajor(Klds + (size_t)(16 * j) * KLD + ks * 32, KLD, lane);
#pragma unroll
      for (int j = 0; j < 4; ++j)
        sc[j] = wmma_bf16(aQ[ks], bK[j], sc[j]);
    }

    // 3) online softmax; butterfly reductions batched step-outer/row-inner
    float tm[8];
#pragma unroll
    for (int r = 0; r < 8; ++r)
      tm[r] = fmaxf(fmaxf(sc[0][r], sc[1][r]), fmaxf(sc[2][r], sc[3][r]));
#pragma unroll
    for (int mk = 8; mk >= 1; mk >>= 1) {
      float sh[8];
#pragma unroll
      for (int r = 0; r < 8; ++r) sh[r] = __shfl_xor(tm[r], mk, 32);
#pragma unroll
      for (int r = 0; r < 8; ++r) tm[r] = fmaxf(tm[r], sh[r]);
    }
    float corr[8], rs[8];
#pragma unroll
    for (int r = 0; r < 8; ++r) {
      float nm = fmaxf(mrow[r], tm[r]);
      corr[r] = __expf(mrow[r] - nm);
      mrow[r] = nm;
      float p0 = __expf(sc[0][r] - nm);
      float p1 = __expf(sc[1][r] - nm);
      float p2 = __expf(sc[2][r] - nm);
      float p3 = __expf(sc[3][r] - nm);
      rs[r] = (p0 + p1) + (p2 + p3);
      int row = (r + 8 * hi) * PLD;
      pbuf[wave][row + nl]      = f2bf(p0);
      pbuf[wave][row + 16 + nl] = f2bf(p1);
      pbuf[wave][row + 32 + nl] = f2bf(p2);
      pbuf[wave][row + 48 + nl] = f2bf(p3);
    }
#pragma unroll
    for (int mk = 8; mk >= 1; mk >>= 1) {
      float sh[8];
#pragma unroll
      for (int r = 0; r < 8; ++r) sh[r] = __shfl_xor(rs[r], mk, 32);
#pragma unroll
      for (int r = 0; r < 8; ++r) rs[r] += sh[r];
    }
#pragma unroll
    for (int r = 0; r < 8; ++r) lrow[r] = lrow[r] * corr[r] + rs[r];

#pragma unroll
    for (int dt = 0; dt < 8; ++dt)
#pragma unroll
      for (int r = 0; r < 8; ++r) o[dt][r] *= corr[r];

    __syncthreads();  // pbuf write -> read; waves aligned (uniform trip count)
    v16bf aP0 = load_a(&pbuf[wave][0], PLD, lane);
    v16bf aP1 = load_a(&pbuf[wave][32], PLD, lane);

    // 4) O += P[16x64] @ V from LDS (16 WMMAs)
#pragma unroll
    for (int g = 0; g < 2; ++g) {
      v16bf bV[4];
#pragma unroll
      for (int j = 0; j < 4; ++j)
        bV[j] = load_b_colmajor(Vlds + (size_t)(g * 64 + 16 * j) * VLD, VLD, lane);
#pragma unroll
      for (int j = 0; j < 4; ++j)
        o[g * 4 + j] = wmma_bf16(aP0, bV[j], o[g * 4 + j]);
#pragma unroll
      for (int j = 0; j < 4; ++j)
        bV[j] = load_b_colmajor(Vlds + (size_t)(g * 64 + 16 * j) * VLD + 32, VLD, lane);
#pragma unroll
      for (int j = 0; j < 4; ++j)
        o[g * 4 + j] = wmma_bf16(aP1, bV[j], o[g * 4 + j]);
    }

    __syncthreads();  // all waves done reading Klds/Vlds
    // 5) drain staged registers into LDS for the next tile
    if (havenext) {
#pragma unroll
      for (int i = 0; i < 8; ++i) {
        int c = tid + i * 128;
        *(v8bf*)(Klds + (size_t)(c >> 3) * KLD + (c & 7) * 8) = kst[i];
      }
#pragma unroll
      for (int i = 0; i < 8; ++i)
        *(v8bf*)(Vlds + (size_t)tid * VLD + i * 8) = vst[i];
    }
    __syncthreads();  // next tile visible
  }

#pragma unroll
  for (int r = 0; r < 8; ++r) lrow[r] = 1.0f / lrow[r];
#pragma unroll
  for (int dt = 0; dt < 8; ++dt) {
#pragma unroll
    for (int r = 0; r < 8; ++r) {
      int s = q0 + r + 8 * hi;
      int col = h * DD + dt * 16 + nl;
      O[((size_t)b * SS + s) * EE + col] = f2bf(o[dt][r] * lrow[r]);
    }
  }
}

extern "C" void kernel_launch(void* const* d_in, const int* in_sizes, int n_in,
                              void* d_out, int out_size, void* d_ws, size_t ws_size,
                              hipStream_t stream) {
  (void)in_sizes; (void)n_in; (void)out_size; (void)ws_size;
  const float* q  = (const float*)d_in[0];
  const float* k  = (const float*)d_in[1];
  const float* v  = (const float*)d_in[2];
  const float* Wq = (const float*)d_in[3];
  const float* Wk = (const float*)d_in[4];
  const float* Wv = (const float*)d_in[5];
  const float* Wo = (const float*)d_in[6];
  float* out = (float*)d_out;

  const size_t ACT = (size_t)MM * EE;  // 8,388,608 elems
  const size_t WEL = (size_t)EE * EE;  // 4,194,304 elems
  char* ws = (char*)d_ws;
  __bf16* qb  = (__bf16*)ws; ws += ACT * 2;
  __bf16* kb  = (__bf16*)ws; ws += ACT * 2;
  __bf16* vb  = (__bf16*)ws; ws += ACT * 2;
  __bf16* Wqb = (__bf16*)ws; ws += WEL * 2;
  __bf16* Wkb = (__bf16*)ws; ws += WEL * 2;
  __bf16* Wvb = (__bf16*)ws; ws += WEL * 2;
  __bf16* Wob = (__bf16*)ws; ws += WEL * 2;
  __bf16* QhB = (__bf16*)ws; ws += ACT * 2;  // [B,H,S,D]
  __bf16* KhB = (__bf16*)ws; ws += ACT * 2;  // [B,H,S,D]
  __bf16* VtB = (__bf16*)ws; ws += ACT * 2;  // [B,H,D,S] (transposed)
  __bf16* Obf = (__bf16*)ws; ws += ACT * 2;  // [B,S,E]

  // fp32 -> bf16
  cvt_bf16<<<(unsigned)((ACT + 255) / 256), 256, 0, stream>>>(q, qb, ACT);
  cvt_bf16<<<(unsigned)((ACT + 255) / 256), 256, 0, stream>>>(k, kb, ACT);
  cvt_bf16<<<(unsigned)((ACT + 255) / 256), 256, 0, stream>>>(v, vb, ACT);
  cvt_bf16<<<(unsigned)((WEL + 255) / 256), 256, 0, stream>>>(Wq, Wqb, WEL);
  cvt_bf16<<<(unsigned)((WEL + 255) / 256), 256, 0, stream>>>(Wk, Wkb, WEL);
  cvt_bf16<<<(unsigned)((WEL + 255) / 256), 256, 0, stream>>>(Wv, Wvb, WEL);
  cvt_bf16<<<(unsigned)((WEL + 255) / 256), 256, 0, stream>>>(Wo, Wob, WEL);

  const float qscale = 0.02209708691207961f;  // 1/sqrt(E): folded into Q projection
  dim3 ggrid(EE / 256, MM / 128);             // (8, 32)
  gemm_qkvo<1><<<ggrid, 256, 0, stream>>>(qb, Wqb, QhB, nullptr, qscale);
  gemm_qkvo<1><<<ggrid, 256, 0, stream>>>(kb, Wkb, KhB, nullptr, 1.0f);
  gemm_qkvo<0><<<ggrid, 256, 0, stream>>>(vb, Wvb, VtB, nullptr, 1.0f);

  attn_fa<<<dim3(SS / 64, BB * HH), 128, 0, stream>>>(QhB, KhB, VtB, Obf);

  gemm_qkvo<2><<<ggrid, 256, 0, stream>>>(Obf, Wob, nullptr, out, 1.0f);
}
